// MambaBlock_87840671137986
// MI455X (gfx1250) — matrix operand order repaired and checked by reading, not compiled, not run on hardware
//
#include <hip/hip_runtime.h>

// ---------------- problem constants (from reference) ----------------
#define DIRS 4
#define LSEQ 128      // sequence length (W or H)
#define CIN  64       // D_MODEL
#define DI   128      // D_INNER
#define NXZ  256      // 2*D_INNER
#define DST  16       // D_STATE
#define DTR  4        // DT_RANK
#define NDBL 48       // DT_RANK + 2*D_STATE (36) padded to 48 for 16-wide tiles
#define NSEQ 256      // sequences per direction (B*H or B*W)
#define NPIX 32768    // B*H*W = 2*128*128

typedef __attribute__((ext_vector_type(16))) __bf16 v16bf;
typedef __attribute__((ext_vector_type(8)))  float  v8f;

__device__ __forceinline__ unsigned short f2bf(float f) {
  unsigned u = __builtin_bit_cast(unsigned, f);
  u += 0x7FFFu + ((u >> 16) & 1u);          // round-to-nearest-even
  return (unsigned short)(u >> 16);
}
__device__ __forceinline__ float bf2f(unsigned short h) {
  unsigned u = ((unsigned)h) << 16;
  return __builtin_bit_cast(float, u);
}
__device__ __forceinline__ __bf16 bfb(unsigned short h) {
  return __builtin_bit_cast(__bf16, h);
}
__device__ __forceinline__ float silu(float x) { return x / (1.f + __expf(-x)); }

// A fragment: 16(M)x32(K) bf16 tile from row-major LDS, base=&S[m0][k0], ld in elems.
// CDNA5 layout: lanes 0-15 & 16-31 both hold M=lane&15; lane-half selects K+8;
// element e -> K = (e&7) + 8*hi + 16*(e>>3).
__device__ __forceinline__ v16bf load_afrag(const unsigned short* base, int ld, int lane) {
  const int lo = lane & 15, hi = lane >> 4;
  v16bf a;
#pragma unroll
  for (int e = 0; e < 16; ++e) {
    const int k = (e & 7) + 8 * hi + 16 * (e >> 3);
    a[e] = bfb(base[lo * ld + k]);
  }
  return a;
}
// B fragment: 32(K)x16(N); B[k][n] = W[n][k]; base=&W[n0][k0], ld in elems.
// Layout: lane holds column N=lane&15; K = (e&7) + 8*hi + 16*(e>>3).
__device__ __forceinline__ v16bf load_bfrag(const unsigned short* base, int ld, int lane) {
  const int lo = lane & 15, hi = lane >> 4;
  v16bf b;
#pragma unroll
  for (int e = 0; e < 16; ++e) b[e] = bfb(base[lo * ld + (e & 7) + 8 * hi + 16 * (e >> 3)]);
  return b;
}
__device__ __forceinline__ v8f wmma_bf16(v16bf a, v16bf b, v8f c) {
  return __builtin_amdgcn_wmma_f32_16x16x32_bf16(false, a, false, b, (short)0, c,
                                                 false, false);
}

// ================== kernel 1: per-(direction, sequence) Mamba ==================
// grid.x = 4*256; block = 256 threads (8 wave32); dynamic LDS = 128 KB.
// LDS regions (bytes):
//   [0,      32768) sX [128][64] bf16  -> later sUC [128][128] bf16
//   [32768,  65536) sW weights (in_w 256x64 / xp_w 48x128 / out_w 64x128) bf16
//   [65536,  98304) sU [128][128] bf16 -> later sDBL [128][48] f32
//   [98304, 131072) sZ [128][128] bf16 -> aliased as sY (read-then-overwrite)
// sU and sZ are contiguous, so the u|z split of GEMM1's 256-wide output is a
// branchless address computation (no divergent stores).
__global__ __launch_bounds__(256) void mamba_dir_kernel(
    const float* __restrict__ x, const float* __restrict__ in_w,
    const float* __restrict__ conv_w, const float* __restrict__ conv_b,
    const float* __restrict__ xp_w, const float* __restrict__ dtp_w,
    const float* __restrict__ dtp_b, const float* __restrict__ A_log,
    const float* __restrict__ Dp, const float* __restrict__ out_w,
    unsigned short* __restrict__ ybuf) {
  extern __shared__ char smem[];
  unsigned short* sX   = (unsigned short*)(smem);
  unsigned short* sUC  = (unsigned short*)(smem);
  unsigned short* sW   = (unsigned short*)(smem + 32768);
  unsigned short* sUZ  = (unsigned short*)(smem + 65536);  // u then z, 16384 elems apart
  unsigned short* sU   = (unsigned short*)(smem + 65536);
  float*          sDBL = (float*)(smem + 65536);
  unsigned short* sZ   = (unsigned short*)(smem + 98304);
  unsigned short* sY   = sZ;  // safe alias: each (t,c) reads z before writing y

  const int tid  = threadIdx.x;
  const int lane = tid & 31;
  const int wid  = __builtin_amdgcn_readfirstlane(tid >> 5);  // scalar wave id
  const int d    = blockIdx.x >> 8;
  const int n    = blockIdx.x & 255;
  const int bimg = n >> 7;     // batch index (H == W == 128)
  const int nrem = n & 127;    // row (d<2) or column (d>=2)

  // ---- Prefetch later-phase weights (global_prefetch_b8) ----
  {
    const float* xp = xp_w + (size_t)d * 36 * DI;
    const float* ow = out_w + (size_t)d * CIN * DI;
    const int line = tid * 32;                 // 32 f32 = one 128B cacheline
    if (line < 36 * DI) __builtin_prefetch(xp + line, 0, 1);
    if (line < CIN * DI) __builtin_prefetch(ow + line, 0, 1);
  }

  // ---- Phase 0: stage in_w[d] and direction-gathered X into LDS (bf16) ----
  {
    const float* W1 = in_w + (size_t)d * NXZ * CIN;
    for (int i = tid; i < NXZ * CIN; i += 256) sW[i] = f2bf(W1[i]);
    for (int i = tid; i < LSEQ * CIN; i += 256) {
      const int c = i >> 7;           // channel
      const int t = i & 127;          // seq position (consecutive -> coalesced)
      int hh, w;
      if (d == 0)      { hh = nrem;     w = t;       }
      else if (d == 1) { hh = nrem;     w = 127 - t; }
      else if (d == 2) { hh = t;        w = nrem;    }
      else             { hh = 127 - t;  w = nrem;    }
      sX[t * CIN + c] = f2bf(x[(((size_t)bimg * CIN + c) * 128 + hh) * 128 + w]);
    }
  }
  __syncthreads();

  // ---- Phase 1 (WMMA): XZ = X(128x64) @ in_w^T(64x256) -> u | z ----
#pragma unroll 2
  for (int tt = 0; tt < 16; ++tt) {
    const int tile = tt * 8 + wid;
    const int mt = tile >> 4, nt = tile & 15;
    v8f acc = {};
#pragma unroll
    for (int kt = 0; kt < 2; ++kt) {
      v16bf a = load_afrag(sX + mt * 16 * CIN + kt * 32, CIN, lane);
      v16bf b = load_bfrag(sW + nt * 16 * CIN + kt * 32, CIN, lane);
      acc = wmma_bf16(a, b, acc);
    }
    const int lo = lane & 15, hi = lane >> 4;
    // branchless u|z destination: z block lives 16384 elems after u block
    const int cbase = (nt & 7) * 16 + lo + ((nt >> 3) << 14);
#pragma unroll
    for (int r = 0; r < 8; ++r) {
      const int row = mt * 16 + r + 8 * hi;
      sUZ[row * DI + cbase] = f2bf(acc[r]);
    }
  }
  __syncthreads();

  // ---- Phase 2: depthwise causal conv (taps=4) + SiLU -> sUC ----
  {
    const float* cw = conv_w + (size_t)d * DI * 4;
    const float* cb = conv_b + (size_t)d * DI;
    for (int i = tid; i < LSEQ * DI; i += 256) {
      const int t = i >> 7, c = i & 127;
      float a = cb[c];
#pragma unroll
      for (int k = 0; k < 4; ++k) {
        const int ts = t - 3 + k;
        if (ts >= 0) a += bf2f(sU[ts * DI + c]) * cw[c * 4 + k];
      }
      sUC[i] = f2bf(silu(a));
    }
  }
  __syncthreads();

  // ---- Phase 3 (WMMA): DBL = uc(128x128) @ xp_w^T(128x36->48) -> f32 LDS ----
  {
    const float* xp = xp_w + (size_t)d * 36 * DI;
    for (int i = tid; i < NDBL * DI; i += 256)
      sW[i] = (i < 36 * DI) ? f2bf(xp[i]) : (unsigned short)0;
  }
  __syncthreads();
#pragma unroll
  for (int tt = 0; tt < 3; ++tt) {
    const int tile = tt * 8 + wid;
    const int mt = tile / 3, nt = tile % 3;
    v8f acc = {};
#pragma unroll
    for (int kt = 0; kt < 4; ++kt) {
      v16bf a = load_afrag(sUC + mt * 16 * DI + kt * 32, DI, lane);
      v16bf b = load_bfrag(sW + nt * 16 * DI + kt * 32, DI, lane);
      acc = wmma_bf16(a, b, acc);
    }
    const int lo = lane & 15, hi = lane >> 4;
#pragma unroll
    for (int r = 0; r < 8; ++r)
      sDBL[(mt * 16 + r + 8 * hi) * NDBL + nt * 16 + lo] = acc[r];
  }
  __syncthreads();

  // ---- Phase 4: selective scan (waves 0..3); out_w preload (waves 4..7) ----
  if (wid < 4) {
    const int c = tid;   // 0..127
    float Ac[DST], hs[DST];
#pragma unroll
    for (int s = 0; s < DST; ++s) {
      Ac[s] = -__expf(A_log[((size_t)d * DI + c) * DST + s]);
      hs[s] = 0.f;
    }
    const float* dw = dtp_w + ((size_t)d * DI + c) * DTR;
    const float w0 = dw[0], w1 = dw[1], w2 = dw[2], w3 = dw[3];
    const float bb = dtp_b[(size_t)d * DI + c];
    const float Dc = Dp[(size_t)d * DI + c];
    for (int t = 0; t < LSEQ; ++t) {
      const float* dr = sDBL + t * NDBL;
      const float xr = dr[0] * w0 + dr[1] * w1 + dr[2] * w2 + dr[3] * w3 + bb;
      const float dt = (xr > 20.f) ? xr : log1pf(__expf(xr));   // softplus
      const float ut = bf2f(sUC[t * DI + c]);
      const float du = dt * ut;
      float y = 0.f;
#pragma unroll
      for (int s = 0; s < DST; ++s) {
        hs[s] = hs[s] * __expf(dt * Ac[s]) + du * dr[DTR + s];  // B_t
        y += hs[s] * dr[DTR + DST + s];                          // C_t
      }
      const float z = bf2f(sZ[t * DI + c]);
      y = (y + ut * Dc) * silu(z);
      sY[t * DI + c] = f2bf(y);   // overwrites z slot after reading it
    }
  } else {
    const float* ow = out_w + (size_t)d * CIN * DI;
    for (int i = tid - 128; i < CIN * DI; i += 128) sW[i] = f2bf(ow[i]);
  }
  __syncthreads();

  // ---- Phase 5 (WMMA): OUT = Y(128x128) @ out_w^T(128x64) -> global (bf16) ----
  unsigned short* yb = ybuf + (size_t)d * NPIX * CIN;   // unscrambled (b,h,w,c)
#pragma unroll
  for (int tt = 0; tt < 4; ++tt) {
    const int tile = tt * 8 + wid;
    const int mt = tile >> 2, nt = tile & 3;
    v8f acc = {};
#pragma unroll
    for (int kt = 0; kt < 4; ++kt) {
      v16bf a = load_afrag(sY + mt * 16 * DI + kt * 32, DI, lane);
      v16bf b = load_bfrag(sW + nt * 16 * DI + kt * 32, DI, lane);
      acc = wmma_bf16(a, b, acc);
    }
    const int lo = lane & 15, hi = lane >> 4;
#pragma unroll
    for (int r = 0; r < 8; ++r) {
      const int t  = mt * 16 + r + 8 * hi;
      const int cm = nt * 16 + lo;
      int hh, w;
      if (d == 0)      { hh = nrem;     w = t;       }
      else if (d == 1) { hh = nrem;     w = 127 - t; }
      else if (d == 2) { hh = t;        w = nrem;    }
      else             { hh = 127 - t;  w = nrem;    }
      const int pix = (bimg * 128 + hh) * 128 + w;
      yb[(size_t)pix * CIN + cm] = f2bf(acc[r]);
    }
  }
}

// ================== kernel 2: fuse + LayerNorm + SiLU + NCHW ==================
// grid.x = 256; 128 pixels per block; dynamic LDS = 128 KB:
//   [0,      65536) sA  [128][256] bf16 (concat of 4 directions)
//   [65536,  98304) sFW [64][256]  bf16
//   [98304, 131072) sO  [128][64]  f32
__global__ __launch_bounds__(256) void fuse_kernel(
    const unsigned short* __restrict__ ybuf, const float* __restrict__ fuse_w,
    const float* __restrict__ fuse_b, const float* __restrict__ ln_g,
    const float* __restrict__ ln_b, float* __restrict__ out) {
  extern __shared__ char smem[];
  unsigned short* sA  = (unsigned short*)(smem);
  unsigned short* sFW = (unsigned short*)(smem + 65536);
  float*          sO  = (float*)(smem + 98304);

  const int tid  = threadIdx.x;
  const int lane = tid & 31;
  const int wid  = __builtin_amdgcn_readfirstlane(tid >> 5);
  const int p0   = blockIdx.x * 128;

  for (int i = tid; i < 128 * 256; i += 256) {
    const int row = i >> 8, col = i & 255;
    const int dd = col >> 6, c = col & 63;
    sA[i] = ybuf[(size_t)dd * NPIX * CIN + (size_t)(p0 + row) * CIN + c];
  }
  for (int i = tid; i < 64 * 256; i += 256) sFW[i] = f2bf(fuse_w[i]);
  __syncthreads();

  // WMMA GEMM: (128 x 64) = A(128x256) @ fuse_w^T(256x64)
#pragma unroll
  for (int tt = 0; tt < 4; ++tt) {
    const int tile = tt * 8 + wid;
    const int mt = tile >> 2, nt = tile & 3;
    v8f acc = {};
#pragma unroll
    for (int kt = 0; kt < 8; ++kt) {
      v16bf a = load_afrag(sA + mt * 16 * 256 + kt * 32, 256, lane);
      v16bf b = load_bfrag(sFW + nt * 16 * 256 + kt * 32, 256, lane);
      acc = wmma_bf16(a, b, acc);
    }
    const int lo = lane & 15, hi = lane >> 4;
#pragma unroll
    for (int r = 0; r < 8; ++r)
      sO[(mt * 16 + r + 8 * hi) * 64 + nt * 16 + lo] = acc[r];
  }
  __syncthreads();

  // bias + LayerNorm(64) + SiLU, store NCHW f32
  if (wid < 4) {
    const int p = p0 + tid;              // tid in 0..127 here
    const int b = p >> 14, hh = (p >> 7) & 127, w = p & 127;
    const float* rowv = sO + tid * 64;
    float mu = 0.f;
    for (int c = 0; c < 64; ++c) mu += rowv[c] + fuse_b[c];
    mu *= (1.f / 64.f);
    float var = 0.f;
    for (int c = 0; c < 64; ++c) {
      const float dv = rowv[c] + fuse_b[c] - mu;
      var += dv * dv;
    }
    var *= (1.f / 64.f);
    const float inv = rsqrtf(var + 1e-5f);
    for (int c = 0; c < 64; ++c) {
      float f = (rowv[c] + fuse_b[c] - mu) * inv * ln_g[c] + ln_b[c];
      f = silu(f);
      out[(((size_t)b * 64 + c) * 128 + hh) * 128 + w] = f;
    }
  }
}

extern "C" void kernel_launch(void* const* d_in, const int* in_sizes, int n_in,
                              void* d_out, int out_size, void* d_ws, size_t ws_size,
                              hipStream_t stream) {
  (void)in_sizes; (void)n_in; (void)out_size; (void)ws_size;
  const float* x      = (const float*)d_in[0];
  const float* in_w   = (const float*)d_in[1];
  const float* conv_w = (const float*)d_in[2];
  const float* conv_b = (const float*)d_in[3];
  const float* xp_w   = (const float*)d_in[4];
  const float* dtp_w  = (const float*)d_in[5];
  const float* dtp_b  = (const float*)d_in[6];
  const float* A_log  = (const float*)d_in[7];
  const float* Dp     = (const float*)d_in[8];
  const float* out_w  = (const float*)d_in[9];
  const float* fuse_w = (const float*)d_in[10];
  const float* fuse_b = (const float*)d_in[11];
  const float* ln_g   = (const float*)d_in[12];
  const float* ln_b   = (const float*)d_in[13];
  float* outp = (float*)d_out;

  // workspace: 4 directions x 32768 pixels x 64 ch of bf16 = 16 MB
  unsigned short* ybuf = (unsigned short*)d_ws;

  mamba_dir_kernel<<<dim3(DIRS * NSEQ), dim3(256), 131072, stream>>>(
      x, in_w, conv_w, conv_b, xp_w, dtp_w, dtp_b, A_log, Dp, out_w, ybuf);
  fuse_kernel<<<dim3(NPIX / 128), dim3(256), 131072, stream>>>(
      ybuf, fuse_w, fuse_b, ln_g, ln_b, outp);
}
